// SelfAttention_86741159510282
// MI455X (gfx1250) — matrix operand order repaired and checked by reading, not compiled, not run on hardware
//
#include <hip/hip_runtime.h>
#include <stdint.h>

#define IN_DIM 1024
#define BS 32
#define NC 1024

typedef __attribute__((ext_vector_type(16))) __bf16 v16bf;
typedef __attribute__((ext_vector_type(8)))  float  v8f;
typedef __attribute__((ext_vector_type(4)))  float  f4;
typedef __attribute__((ext_vector_type(4)))  unsigned int u4;

union FragB { v16bf v; u4 q[2]; };

__device__ __forceinline__ unsigned short f2bf(float f) {
    unsigned u = __float_as_uint(f);
    u += 0x7FFFu + ((u >> 16) & 1u);   // round-to-nearest-even
    return (unsigned short)(u >> 16);
}
__device__ __forceinline__ unsigned pk2(float lo, float hi) {
    return (unsigned)f2bf(lo) | ((unsigned)f2bf(hi) << 16);
}

// convert 16 f32 -> 16 bf16 and store as two b128 into an LDS row
__device__ __forceinline__ void cvt_store16(unsigned short (*dst)[40], int r, int h,
                                            f4 f0, f4 f1, f4 f2, f4 f3) {
    u4 u0 = {pk2(f0.x, f0.y), pk2(f0.z, f0.w), pk2(f1.x, f1.y), pk2(f1.z, f1.w)};
    u4 u1 = {pk2(f2.x, f2.y), pk2(f2.z, f2.w), pk2(f3.x, f3.y), pk2(f3.z, f3.w)};
    *(u4*)&dst[r][h * 16]     = u0;
    *(u4*)&dst[r][h * 16 + 8] = u1;
}

// ---------------------------------------------------------------------------
// Kernel 1: q[b,n] = x[b,n,:]·Wq + bq ; k likewise. One wave32 per row.
// ---------------------------------------------------------------------------
__global__ __launch_bounds__(256) void qk_kernel(
    const float* __restrict__ x, const float* __restrict__ Wq,
    const float* __restrict__ bq, const float* __restrict__ Wk,
    const float* __restrict__ bk, float* __restrict__ q, float* __restrict__ k) {
    int wave = threadIdx.x >> 5, lane = threadIdx.x & 31;
    int row = blockIdx.x * 8 + wave;               // 0..32767 = b*1024+n
    const float* xr = x + (size_t)row * IN_DIM;
    float aq = 0.f, ak = 0.f;
#pragma unroll
    for (int i = 0; i < 8; ++i) {
        int off = lane * 4 + i * 128;
        f4 xv = *(const f4*)(xr + off);
        f4 wq = *(const f4*)(Wq + off);
        f4 wk = *(const f4*)(Wk + off);
        aq += xv.x * wq.x + xv.y * wq.y + xv.z * wq.z + xv.w * wq.w;
        ak += xv.x * wk.x + xv.y * wk.y + xv.z * wk.z + xv.w * wk.w;
    }
#pragma unroll
    for (int s = 16; s > 0; s >>= 1) {
        aq += __shfl_xor(aq, s, 32);
        ak += __shfl_xor(ak, s, 32);
    }
    if (lane == 0) { q[row] = aq + bq[0]; k[row] = ak + bk[0]; }
}

// ---------------------------------------------------------------------------
// Kernel 2: stable-softmax normalizers.  energy[b,n,m] = q[b,n]*k[b,m].
// shift c_n = max(q_n*kmax, q_n*kmin);  rz_n = 1/sum_m exp(q_n*k_m - c_n).
// ---------------------------------------------------------------------------
__global__ __launch_bounds__(256) void z_kernel(
    const float* __restrict__ q, const float* __restrict__ k,
    float* __restrict__ sh, float* __restrict__ rz) {
    __shared__ float ks[1024];
    __shared__ float rmx[256], rmn[256];
    int b = blockIdx.x, t = threadIdx.x;
    float mx = -1e30f, mn = 1e30f;
#pragma unroll
    for (int i = 0; i < 4; ++i) {
        float v = k[b * 1024 + t + 256 * i];
        ks[t + 256 * i] = v;
        mx = fmaxf(mx, v); mn = fminf(mn, v);
    }
    rmx[t] = mx; rmn[t] = mn;
    __syncthreads();
    for (int s = 128; s > 0; s >>= 1) {
        if (t < s) { rmx[t] = fmaxf(rmx[t], rmx[t + s]); rmn[t] = fminf(rmn[t], rmn[t + s]); }
        __syncthreads();
    }
    float kmx = rmx[0], kmn = rmn[0];
    for (int i = 0; i < 4; ++i) {
        int n = t + 256 * i;
        float qn = q[b * 1024 + n];
        float c = fmaxf(qn * kmx, qn * kmn);
        float acc = 0.f;
        for (int m = 0; m < 1024; ++m) acc += __expf(qn * ks[m] - c);
        sh[b * 1024 + n] = c;
        rz[b * 1024 + n] = 1.f / acc;
    }
}

// ---------------------------------------------------------------------------
// Kernel 3: attention[b,n,m] = exp(q_n*k_m - c_n) * rz_n  -> d_out 2nd half.
// ---------------------------------------------------------------------------
__global__ __launch_bounds__(256) void attn_kernel(
    const float* __restrict__ q, const float* __restrict__ k,
    const float* __restrict__ sh, const float* __restrict__ rz,
    float* __restrict__ attn) {
    size_t gid = (size_t)blockIdx.x * 256 + threadIdx.x;   // 8M threads, 4 elems each
    int m4 = (int)(gid & 255);
    int row = (int)(gid >> 8);                             // b*1024+n
    int b = row >> 10;
    float qn = q[row], c = sh[row], r = rz[row];
    f4 kv = *(const f4*)(k + b * 1024 + m4 * 4);
    f4 o;
    o.x = __expf(qn * kv.x - c) * r;
    o.y = __expf(qn * kv.y - c) * r;
    o.z = __expf(qn * kv.z - c) * r;
    o.w = __expf(qn * kv.w - c) * r;
    *(f4*)(attn + ((size_t)row << 10) + m4 * 4) = o;
}

// ---------------------------------------------------------------------------
// WMMA GEMM 1:  V[b] = bf16( Wv(1024x1024) * X[b]^T + bv )   (M=o, N=n, K=d)
// 128x128 block tile, K-step 32 (== WMMA K), 8 waves each 32x64 (2x4 tiles).
// Double-buffered LDS: prefetch tile k+1 into regs before WMMA burst on k,
// one barrier per K-iteration.
// ---------------------------------------------------------------------------
__global__ __launch_bounds__(256) void gemm1_kernel(
    const float* __restrict__ Wv, const float* __restrict__ x,
    const float* __restrict__ bv, unsigned short* __restrict__ V16) {
    __shared__ __align__(16) unsigned short Al[2][128][40];   // [m][k] padded rows
    __shared__ __align__(16) unsigned short Bl[2][128][40];   // [n][k]
    __shared__ float bvs[128];
    int tid = threadIdx.x;
    int blkm = blockIdx.x >> 3, blkn = blockIdx.x & 7, b = blockIdx.y;
    if (tid < 128) bvs[tid] = bv[blkm * 128 + tid];
    int wave = tid >> 5, lane = tid & 31;
    int lm = lane & 15, hi = lane >> 4;
    int m_off = (wave & 3) * 32, n_off = (wave >> 2) * 64;

    v8f acc[2][4];
    v8f zero = {0.f, 0.f, 0.f, 0.f, 0.f, 0.f, 0.f, 0.f};
#pragma unroll
    for (int i = 0; i < 2; ++i)
#pragma unroll
        for (int j = 0; j < 4; ++j) acc[i][j] = zero;

    int r = tid >> 1, h = tid & 1;
    const float* asrc = Wv + (size_t)(blkm * 128 + r) * 1024 + h * 16;
    const float* bsrc = x + ((size_t)b << 20) + (size_t)(blkn * 128 + r) * 1024 + h * 16;

    // prologue: load + stage tile 0 into buffer 0
    f4 a0, a1, a2, a3, b0, b1, b2, b3;
    {
        const f4* s = (const f4*)asrc;
        a0 = s[0]; a1 = s[1]; a2 = s[2]; a3 = s[3];
        s = (const f4*)bsrc;
        b0 = s[0]; b1 = s[1]; b2 = s[2]; b3 = s[3];
    }
    cvt_store16(Al[0], r, h, a0, a1, a2, a3);
    cvt_store16(Bl[0], r, h, b0, b1, b2, b3);
    __syncthreads();

    int buf = 0;
    for (int k0 = 0; k0 < 1024; k0 += 32) {
        bool more = (k0 + 32) < 1024;
        if (more) {   // prefetch next tile (issues before the WMMA burst)
            const f4* s = (const f4*)(asrc + k0 + 32);
            a0 = s[0]; a1 = s[1]; a2 = s[2]; a3 = s[3];
            s = (const f4*)(bsrc + k0 + 32);
            b0 = s[0]; b1 = s[1]; b2 = s[2]; b3 = s[3];
        }
        unsigned short (*Ac)[40] = Al[buf];
        unsigned short (*Bc)[40] = Bl[buf];
        FragB a[2], bb[4];
#pragma unroll
        for (int sm = 0; sm < 2; ++sm) {
            int row = m_off + sm * 16 + lm, ka = hi ? 8 : 0;
            a[sm].q[0] = *(const u4*)&Ac[row][ka];
            a[sm].q[1] = *(const u4*)&Ac[row][16 + ka];
        }
#pragma unroll
        for (int sn = 0; sn < 4; ++sn) {
            int col = n_off + sn * 16 + lm, kb = hi ? 8 : 0;
            bb[sn].q[0] = *(const u4*)&Bc[col][kb];
            bb[sn].q[1] = *(const u4*)&Bc[col][16 + kb];
        }
#pragma unroll
        for (int sm = 0; sm < 2; ++sm)
#pragma unroll
            for (int sn = 0; sn < 4; ++sn)
                acc[sm][sn] = __builtin_amdgcn_wmma_f32_16x16x32_bf16(
                    false, a[sm].v, false, bb[sn].v, (short)0, acc[sm][sn], false, false);
        if (more) {
            cvt_store16(Al[buf ^ 1], r, h, a0, a1, a2, a3);
            cvt_store16(Bl[buf ^ 1], r, h, b0, b1, b2, b3);
            buf ^= 1;
        }
        __syncthreads();   // one barrier per K-iteration
    }
#pragma unroll
    for (int sm = 0; sm < 2; ++sm)
#pragma unroll
        for (int sn = 0; sn < 4; ++sn)
#pragma unroll
            for (int v = 0; v < 8; ++v) {
                int ml = m_off + sm * 16 + hi * 8 + v;
                int m_g = blkm * 128 + ml;
                int n_g = blkn * 128 + n_off + sn * 16 + lm;
                V16[((size_t)b << 20) + (size_t)m_g * 1024 + n_g] =
                    f2bf(acc[sm][sn][v] + bvs[ml]);
            }
}

// ---------------------------------------------------------------------------
// WMMA GEMM 2:  out[b] = V[b](1024x1024,bf16) * attention[b]  (M=o, N=m, K=n)
// A: bf16 copy-through. B: attention f32 row-major [k][m] -> LDS [m][k].
// Same double-buffer scheme as GEMM 1.
// ---------------------------------------------------------------------------
__global__ __launch_bounds__(256) void gemm2_kernel(
    const unsigned short* __restrict__ V16, const float* __restrict__ attn,
    float* __restrict__ out) {
    __shared__ __align__(16) unsigned short Al[2][128][40];
    __shared__ __align__(16) unsigned short Bl[2][128][40];
    int tid = threadIdx.x;
    int blkm = blockIdx.x >> 3, blkn = blockIdx.x & 7, b = blockIdx.y;
    int wave = tid >> 5, lane = tid & 31;
    int lm = lane & 15, hi = lane >> 4;
    int m_off = (wave & 3) * 32, n_off = (wave >> 2) * 64;

    v8f acc[2][4];
    v8f zero = {0.f, 0.f, 0.f, 0.f, 0.f, 0.f, 0.f, 0.f};
#pragma unroll
    for (int i = 0; i < 2; ++i)
#pragma unroll
        for (int j = 0; j < 4; ++j) acc[i][j] = zero;

    int r = tid >> 1, h = tid & 1;
    const unsigned short* asrc =
        V16 + ((size_t)b << 20) + (size_t)(blkm * 128 + r) * 1024 + h * 16;
    int kk = tid >> 3, mg = tid & 7;
    const float* bsrc =
        attn + ((size_t)b << 20) + (size_t)kk * 1024 + blkn * 128 + mg * 16;

    // prologue: tile 0 into buffer 0
    u4 av0, av1;
    f4 t0, t1, t2, t3;
    {
        const u4* s = (const u4*)asrc;
        av0 = s[0]; av1 = s[1];
        const f4* bs = (const f4*)bsrc;
        t0 = bs[0]; t1 = bs[1]; t2 = bs[2]; t3 = bs[3];
    }
    *(u4*)&Al[0][r][h * 16]     = av0;
    *(u4*)&Al[0][r][h * 16 + 8] = av1;
    {
        int mb = mg * 16;
        Bl[0][mb + 0][kk] = f2bf(t0.x);  Bl[0][mb + 1][kk] = f2bf(t0.y);
        Bl[0][mb + 2][kk] = f2bf(t0.z);  Bl[0][mb + 3][kk] = f2bf(t0.w);
        Bl[0][mb + 4][kk] = f2bf(t1.x);  Bl[0][mb + 5][kk] = f2bf(t1.y);
        Bl[0][mb + 6][kk] = f2bf(t1.z);  Bl[0][mb + 7][kk] = f2bf(t1.w);
        Bl[0][mb + 8][kk] = f2bf(t2.x);  Bl[0][mb + 9][kk] = f2bf(t2.y);
        Bl[0][mb +10][kk] = f2bf(t2.z);  Bl[0][mb +11][kk] = f2bf(t2.w);
        Bl[0][mb +12][kk] = f2bf(t3.x);  Bl[0][mb +13][kk] = f2bf(t3.y);
        Bl[0][mb +14][kk] = f2bf(t3.z);  Bl[0][mb +15][kk] = f2bf(t3.w);
    }
    __syncthreads();

    int buf = 0;
    for (int k0 = 0; k0 < 1024; k0 += 32) {
        bool more = (k0 + 32) < 1024;
        if (more) {   // prefetch next tile
            const u4* s = (const u4*)(asrc + k0 + 32);
            av0 = s[0]; av1 = s[1];
            const f4* bs = (const f4*)(bsrc + (size_t)(k0 + 32) * 1024);
            t0 = bs[0]; t1 = bs[1]; t2 = bs[2]; t3 = bs[3];
        }
        unsigned short (*Ac)[40] = Al[buf];
        unsigned short (*Bc)[40] = Bl[buf];
        FragB a[2], bb[4];
#pragma unroll
        for (int sm = 0; sm < 2; ++sm) {
            int row = m_off + sm * 16 + lm, ka = hi ? 8 : 0;
            a[sm].q[0] = *(const u4*)&Ac[row][ka];
            a[sm].q[1] = *(const u4*)&Ac[row][16 + ka];
        }
#pragma unroll
        for (int sn = 0; sn < 4; ++sn) {
            int col = n_off + sn * 16 + lm, kb = hi ? 8 : 0;
            bb[sn].q[0] = *(const u4*)&Bc[col][kb];
            bb[sn].q[1] = *(const u4*)&Bc[col][16 + kb];
        }
#pragma unroll
        for (int sm = 0; sm < 2; ++sm)
#pragma unroll
            for (int sn = 0; sn < 4; ++sn)
                acc[sm][sn] = __builtin_amdgcn_wmma_f32_16x16x32_bf16(
                    false, a[sm].v, false, bb[sn].v, (short)0, acc[sm][sn], false, false);
        if (more) {
            unsigned short (*An)[40] = Al[buf ^ 1];
            unsigned short (*Bn)[40] = Bl[buf ^ 1];
            *(u4*)&An[r][h * 16]     = av0;
            *(u4*)&An[r][h * 16 + 8] = av1;
            int mb = mg * 16;
            Bn[mb + 0][kk] = f2bf(t0.x);  Bn[mb + 1][kk] = f2bf(t0.y);
            Bn[mb + 2][kk] = f2bf(t0.z);  Bn[mb + 3][kk] = f2bf(t0.w);
            Bn[mb + 4][kk] = f2bf(t1.x);  Bn[mb + 5][kk] = f2bf(t1.y);
            Bn[mb + 6][kk] = f2bf(t1.z);  Bn[mb + 7][kk] = f2bf(t1.w);
            Bn[mb + 8][kk] = f2bf(t2.x);  Bn[mb + 9][kk] = f2bf(t2.y);
            Bn[mb +10][kk] = f2bf(t2.z);  Bn[mb +11][kk] = f2bf(t2.w);
            Bn[mb +12][kk] = f2bf(t3.x);  Bn[mb +13][kk] = f2bf(t3.y);
            Bn[mb +14][kk] = f2bf(t3.z);  Bn[mb +15][kk] = f2bf(t3.w);
            buf ^= 1;
        }
        __syncthreads();
    }
#pragma unroll
    for (int sm = 0; sm < 2; ++sm)
#pragma unroll
        for (int sn = 0; sn < 4; ++sn)
#pragma unroll
            for (int v = 0; v < 8; ++v) {
                int m_g = blkm * 128 + m_off + sm * 16 + hi * 8 + v;
                int n_g = blkn * 128 + n_off + sn * 16 + lm;
                out[((size_t)b << 20) + (size_t)m_g * 1024 + n_g] = acc[sm][sn][v];
            }
}

// ---------------------------------------------------------------------------
extern "C" void kernel_launch(void* const* d_in, const int* in_sizes, int n_in,
                              void* d_out, int out_size, void* d_ws, size_t ws_size,
                              hipStream_t stream) {
    const float* x  = (const float*)d_in[0];
    const float* Wq = (const float*)d_in[1];
    const float* bq = (const float*)d_in[2];
    const float* Wk = (const float*)d_in[3];
    const float* bk = (const float*)d_in[4];
    const float* Wv = (const float*)d_in[5];
    const float* bv = (const float*)d_in[6];

    float* out  = (float*)d_out;                       // (bs, d, n) flat == (bs,n,d) view
    float* attn = out + (size_t)BS * NC * IN_DIM;      // second output

    char* w = (char*)d_ws;
    float* q  = (float*)(w);                           // 128 KB
    float* k  = (float*)(w + (128 << 10));             // 128 KB
    float* sh = (float*)(w + (256 << 10));             // 128 KB
    float* rz = (float*)(w + (384 << 10));             // 128 KB
    unsigned short* V16 = (unsigned short*)(w + (512 << 10));  // 64 MB bf16 V

    qk_kernel<<<4096, 256, 0, stream>>>(x, Wq, bq, Wk, bk, q, k);
    z_kernel<<<BS, 256, 0, stream>>>(q, k, sh, rz);
    attn_kernel<<<32768, 256, 0, stream>>>(q, k, sh, rz, attn);
    gemm1_kernel<<<dim3(64, BS), 256, 0, stream>>>(Wv, x, bv, V16);
    gemm2_kernel<<<dim3(64, BS), 256, 0, stream>>>(V16, attn, out);
}
